// Int4QuantLinear_13048110645378
// MI455X (gfx1250) — compile-verified, matrix operand order checked by reading, not run on hardware
//
#include <hip/hip_runtime.h>

// ---------------------------------------------------------------------------
// Int4 quant linear + hp correction on MI455X (gfx1250), f16 WMMA GEMM.
// K augmented 4096 -> 4224 so the hp gather term is part of the same GEMM.
// Memory-bound (~575 MB HBM vs 0.57 TFLOP): one pass over x (f32->f16 on the
// fly), W_eff resident in L2, double-buffered LDS, BK=96 (3 WMMA substeps),
// B-tile copied with CDNA5 async global->LDS (ASYNCcnt-tracked).
// ---------------------------------------------------------------------------

typedef __attribute__((ext_vector_type(16))) _Float16 v16h;
typedef __attribute__((ext_vector_type(8)))  _Float16 v8h;
typedef __attribute__((ext_vector_type(8)))  float    v8f;

#ifndef __has_builtin
#define __has_builtin(x) 0
#endif

#if __has_builtin(__builtin_amdgcn_global_load_async_to_lds_b128)
#define USE_ASYNC_B 1
typedef int v4i __attribute__((ext_vector_type(4)));
typedef v4i __attribute__((address_space(1)))* v4i_gp;   // global (prints as __device__)
typedef v4i __attribute__((address_space(3)))* v4i_lp;   // LDS
#else
#define USE_ASYNC_B 0
#endif

__device__ __forceinline__ void wait_async_zero() {
#if __has_builtin(__builtin_amdgcn_s_wait_asynccnt)
    __builtin_amdgcn_s_wait_asynccnt(0);
#else
    asm volatile("s_wait_asynccnt 0x0" ::: "memory");
#endif
}

#define IN_F     4096
#define OUT_F    4096
#define K_AUG    4224          // 4096 + 128 hp columns
#define NHP      128
#define NGRP     32
#define M_TOTAL  16384         // 8 * 2048 tokens

#define BM       128
#define BN       128
#define BK       96            // 3 x K32 substeps; 4224/96 = 44 iterations
#define LDSK     104           // BK + 8 halves pad -> 208B stride, conflict-free
#define ABUF     (BM * LDSK)
#define BBUF     (BN * LDSK)

// ---------------------------------------------------------------------------
// Kernel 1: build W_eff (f16, [OUT_F][K_AUG]) in workspace.
// ---------------------------------------------------------------------------
__global__ __launch_bounds__(256) void dequant_pack(
    const int*   __restrict__ qw,      // [OUT_F][2048]
    const float* __restrict__ scales,  // [OUT_F][32]
    const float* __restrict__ zeros,   // [OUT_F][32]
    const float* __restrict__ hpw,     // [OUT_F][128]
    _Float16*    __restrict__ weff)    // [OUT_F][K_AUG]
{
    int gid = blockIdx.x * 256 + threadIdx.x;   // 0 .. OUT_F*2112-1
    int o = gid / (K_AUG / 2);
    int j = gid - o * (K_AUG / 2);              // pair index within row
    float w0, w1;
    if (j < IN_F / 2) {
        int q = qw[o * (IN_F / 2) + j];
        int g = j >> 6;                          // group = (2j)/128
        float s = fmaxf(scales[o * NGRP + g], 1e-9f);
        float z = zeros[o * NGRP + g];
        w0 = ((float)(q & 15)        - z) * s;   // element 2j   (low nibble)
        w1 = ((float)((q >> 4) & 15) - z) * s;   // element 2j+1 (high nibble)
    } else {
        int h = (j - IN_F / 2) * 2;
        w0 = hpw[o * NHP + h];
        w1 = hpw[o * NHP + h + 1];
    }
    union { _Float16 h[2]; unsigned u; } pk;
    pk.h[0] = (_Float16)w0;
    pk.h[1] = (_Float16)w1;
    ((unsigned*)weff)[(size_t)o * (K_AUG / 2) + j] = pk.u;
}

// ---------------------------------------------------------------------------
// Kernel 2: tiled WMMA GEMM, double-buffered LDS, phase-interleaved loads.
// ---------------------------------------------------------------------------
__global__ __launch_bounds__(256, 1) void gemm_wmma(
    const float*    __restrict__ x,     // [M_TOTAL][IN_F] f32
    const _Float16* __restrict__ weff,  // [OUT_F][K_AUG]  f16
    const float*    __restrict__ bias,  // [OUT_F]
    const int*      __restrict__ hpidx, // [NHP]
    float*          __restrict__ out)   // [M_TOTAL][OUT_F]
{
    __shared__ __align__(16) _Float16 lds_a[2 * ABUF];
    __shared__ __align__(16) _Float16 lds_b[2 * BBUF];
    __shared__ int sidx[NHP];

    const int tid  = threadIdx.x;
    const int lane = tid & 31;
    const int wid  = tid >> 5;          // 0..7
    const int wm   = wid >> 2;          // 0..1  -> 64 output rows per wave
    const int wn   = wid & 3;           // 0..3  -> 32 output cols per wave
    const int block_m = blockIdx.y * BM;
    const int block_n = blockIdx.x * BN;

    if (tid < NHP) sidx[tid] = hpidx[tid];

    // A: 128 rows x 24 float4-chunks = 3072 chunks -> 12 per thread.
    // B: 128 rows x 12 uint4-chunks  = 1536 chunks -> 6 per thread.
    auto loadA = [&](int kb, int i, float4& f, bool tail) {
        const int c   = tid + 256 * i;
        const int row = c / 24;
        const int col = (c - row * 24) * 4;
        const int gk  = kb + col;
        const float* xr = x + (size_t)(block_m + row) * IN_F;
        if (!tail) {                               // uniform fast path
            f = *(const float4*)(xr + gk);
        } else if (gk < IN_F) {
            f = *(const float4*)(xr + gk);
        } else {                                   // hp gather columns
            f.x = xr[sidx[gk - IN_F + 0]];
            f.y = xr[sidx[gk - IN_F + 1]];
            f.z = xr[sidx[gk - IN_F + 2]];
            f.w = xr[sidx[gk - IN_F + 3]];
        }
    };
    auto storeA = [&](int buf, int i, const float4& f) {
        const int c   = tid + 256 * i;
        const int row = c / 24;
        const int col = (c - row * 24) * 4;
        union { _Float16 h[4]; uint2 u; } pk;
        pk.h[0] = (_Float16)f.x;
        pk.h[1] = (_Float16)f.y;
        pk.h[2] = (_Float16)f.z;
        pk.h[3] = (_Float16)f.w;
        *(uint2*)&lds_a[buf * ABUF + row * LDSK + col] = pk.u;
    };

#if USE_ASYNC_B
    auto asyncB = [&](int kb, int buf) {
#pragma unroll
        for (int i = 0; i < 6; ++i) {
            const int c   = tid + 256 * i;
            const int row = c / 12;
            const int col = (c - row * 12) * 8;
            const _Float16* gp = weff + (size_t)(block_n + row) * K_AUG + kb + col;
            _Float16* lp = &lds_b[buf * BBUF + row * LDSK + col];
            __builtin_amdgcn_global_load_async_to_lds_b128(
                (v4i_gp)gp, (v4i_lp)lp, /*imm offset*/0, /*cpol*/0);
        }
    };
#else
    auto loadB = [&](int kb, int i, uint4& u) {
        const int c   = tid + 256 * i;
        const int row = c / 12;
        const int col = (c - row * 12) * 8;
        u = *(const uint4*)(weff + (size_t)(block_n + row) * K_AUG + kb + col);
    };
    auto storeB = [&](int buf, int i, const uint4& u) {
        const int c   = tid + 256 * i;
        const int row = c / 12;
        const int col = (c - row * 12) * 8;
        *(uint4*)&lds_b[buf * BBUF + row * LDSK + col] = u;
    };
#endif

    // ISA A/B fragment layout (16-bit, 16x32): per lane 8 halves at K-offset
    // (lane>=16 ? 8 : 0), plus 8 more at +16 halves.
    const int kb_off = (lane >> 4) * 8;
    const int fr_row = lane & 15;

    auto load_frag = [&](const _Float16* base) -> v16h {
        union { v16h v; v8h h[2]; } f;
        f.h[0] = *(const v8h*)(base);
        f.h[1] = *(const v8h*)(base + 16);
        return f.v;
    };

    // --- prologue: fill buffer 0 ---
    {
        float4 fa[12];
#pragma unroll
        for (int i = 0; i < 12; ++i) loadA(0, i, fa[i], false);
#if USE_ASYNC_B
        asyncB(0, 0);
#else
        uint4 fb[6];
#pragma unroll
        for (int i = 0; i < 6; ++i)  loadB(0, i, fb[i]);
#endif
#pragma unroll
        for (int i = 0; i < 12; ++i) storeA(0, i, fa[i]);
#if USE_ASYNC_B
        wait_async_zero();
#else
#pragma unroll
        for (int i = 0; i < 6; ++i)  storeB(0, i, fb[i]);
#endif
    }

    v8f acc[4][2];
#pragma unroll
    for (int mt = 0; mt < 4; ++mt)
#pragma unroll
        for (int nt = 0; nt < 2; ++nt)
            acc[mt][nt] = (v8f)0.0f;

    auto compute_ks = [&](int buf, int ks) {
        const _Float16* la = &lds_a[buf * ABUF];
        const _Float16* lb = &lds_b[buf * BBUF];
        v16h bfrag[2];
#pragma unroll
        for (int nt = 0; nt < 2; ++nt) {
            const int nl = wn * 32 + nt * 16 + fr_row;
            bfrag[nt] = load_frag(&lb[nl * LDSK + ks + kb_off]);
        }
#pragma unroll
        for (int mt = 0; mt < 4; ++mt) {
            const int ml = wm * 64 + mt * 16 + fr_row;
            v16h afrag = load_frag(&la[ml * LDSK + ks + kb_off]);
#pragma unroll
            for (int nt = 0; nt < 2; ++nt) {
                acc[mt][nt] = __builtin_amdgcn_wmma_f32_16x16x32_f16(
                    false, afrag, false, bfrag[nt],
                    (short)0, acc[mt][nt], false, false);
            }
        }
    };

    __syncthreads();

    const int KITERS = K_AUG / BK;   // 44
    for (int it = 0; it < KITERS; ++it) {
        const int  buf      = it & 1;
        const int  nbuf     = buf ^ 1;
        const int  kb_next  = (it + 1) * BK;
        const bool has_next = (it + 1 < KITERS);
        const bool tail     = (kb_next + BK > IN_F);  // uniform: gather possible

        float4 fa[6];
#if !USE_ASYNC_B
        uint4  fb[6];
#endif

        if (has_next) {
#if USE_ASYNC_B
            asyncB(kb_next, nbuf);       // fire-and-forget, no VGPR staging
#endif
#pragma unroll
            for (int i = 0; i < 6; ++i) loadA(kb_next, i, fa[i], tail);
        }
        compute_ks(buf, 0);
        if (has_next) {
#pragma unroll
            for (int i = 0; i < 6; ++i) storeA(nbuf, i, fa[i]);
#pragma unroll
            for (int i = 0; i < 6; ++i) loadA(kb_next, 6 + i, fa[i], tail);
        }
        compute_ks(buf, 32);
        if (has_next) {
#pragma unroll
            for (int i = 0; i < 6; ++i) storeA(nbuf, 6 + i, fa[i]);
#if !USE_ASYNC_B
#pragma unroll
            for (int i = 0; i < 6; ++i) loadB(kb_next, i, fb[i]);
#endif
        }
        compute_ks(buf, 64);
#if USE_ASYNC_B
        wait_async_zero();               // LDS writes visible before barrier
#else
        if (has_next) {
#pragma unroll
            for (int i = 0; i < 6; ++i) storeB(nbuf, i, fb[i]);
        }
#endif
        __syncthreads();
    }

    // Epilogue: C layout -> VGPR r holds M=r (lanes 0-15) / M=r+8 (lanes 16-31),
    // N = lane & 15. Add bias, store f32.
#pragma unroll
    for (int nt = 0; nt < 2; ++nt) {
        const int n  = block_n + wn * 32 + nt * 16 + fr_row;
        const float bv = bias[n];
#pragma unroll
        for (int mt = 0; mt < 4; ++mt) {
            const int m_base = block_m + wm * 64 + mt * 16 + ((lane >> 4) << 3);
#pragma unroll
            for (int r = 0; r < 8; ++r) {
                out[(size_t)(m_base + r) * OUT_F + n] = acc[mt][nt][r] + bv;
            }
        }
    }
}

// ---------------------------------------------------------------------------
extern "C" void kernel_launch(void* const* d_in, const int* in_sizes, int n_in,
                              void* d_out, int out_size, void* d_ws, size_t ws_size,
                              hipStream_t stream) {
    const float* x          = (const float*)d_in[0];
    const int*   qweight    = (const int*)  d_in[1];
    const float* scales     = (const float*)d_in[2];
    const float* zeros      = (const float*)d_in[3];
    const float* bias       = (const float*)d_in[4];
    const int*   hp_indices = (const int*)  d_in[5];
    const float* hp_weight  = (const float*)d_in[6];
    float*       out        = (float*)d_out;
    _Float16*    weff       = (_Float16*)d_ws;   // OUT_F * K_AUG halves = 33 MB

    dequant_pack<<<(OUT_F * (K_AUG / 2)) / 256, 256, 0, stream>>>(
        qweight, scales, zeros, hp_weight, weff);

    dim3 grid(OUT_F / BN, M_TOTAL / BM);   // N fast-varying -> x reused in L2
    gemm_wmma<<<grid, 256, 0, stream>>>(x, weff, bias, hp_indices, out);
}
